// PagedGraniteBlock_7138235646728
// MI455X (gfx1250) — compile-verified, matrix operand order checked by reading, not run on hardware
//
#include <hip/hip_runtime.h>

// ---------------- constants (Granite block, B=1) ----------------
#define SEQ   2048
#define DIM   4096
#define NH    32
#define HD    128
#define HID   11008
#define EPSF  1e-5f
#define SCALEF 0.08838834764831845f   // 1/sqrt(128)

// ---------------- CDNA5 types ----------------
typedef __attribute__((ext_vector_type(16))) __bf16       v16bf;
typedef __attribute__((ext_vector_type(8)))  float        v8f;
typedef __attribute__((ext_vector_type(4)))  unsigned int v4u;
typedef __attribute__((ext_vector_type(4)))  float        v4f;

union FragB16 { v16bf v; v4u q[2]; unsigned short u[16]; };

__device__ __forceinline__ unsigned short f32_to_bf16(float f) {
  unsigned int u = __float_as_uint(f);
  unsigned int r = u + 0x7FFFu + ((u >> 16) & 1u);   // round-to-nearest-even
  return (unsigned short)(r >> 16);
}

// =================================================================
// One-time weight prep: B[K,N] f32 -> Bt[N,K] bf16 (64x64 LDS tiles)
// Paid once per launch; makes every GEMM's B operand bf16 and
// L2-resident (33-90MB << 192MB) across the 16 M-block sweeps.
// =================================================================
__global__ __launch_bounds__(256)
void transpose_cvt_kernel(const float* __restrict__ B, unsigned short* __restrict__ Bt,
                          int K, int N) {
  __shared__ float tile[64][65];
  const int t  = threadIdx.x;
  const int k0 = blockIdx.y * 64;
  const int n0 = blockIdx.x * 64;
  const int r  = t >> 4;          // 0..15
  const int c4 = (t & 15) * 4;    // 0..60
#pragma unroll
  for (int i = 0; i < 4; ++i) {
    int row = r + 16 * i;
    v4f v = *(const v4f*)(B + (size_t)(k0 + row) * N + n0 + c4);
    tile[row][c4 + 0] = v[0]; tile[row][c4 + 1] = v[1];
    tile[row][c4 + 2] = v[2]; tile[row][c4 + 3] = v[3];
  }
  __syncthreads();
#pragma unroll
  for (int i = 0; i < 4; ++i) {
    int n = r + 16 * i;
    unsigned short p0 = f32_to_bf16(tile[c4 + 0][n]);
    unsigned short p1 = f32_to_bf16(tile[c4 + 1][n]);
    unsigned short p2 = f32_to_bf16(tile[c4 + 2][n]);
    unsigned short p3 = f32_to_bf16(tile[c4 + 3][n]);
    unsigned long long pk = (unsigned long long)p0 | ((unsigned long long)p1 << 16) |
                            ((unsigned long long)p2 << 32) | ((unsigned long long)p3 << 48);
    *(unsigned long long*)(Bt + (size_t)(n0 + n) * K + k0 + c4) = pk;
  }
}

// =================================================================
// RMSNorm: per-row rsqrt(mean(x^2)+eps), output bf16 activations
// =================================================================
__global__ __launch_bounds__(256)
void rmsnorm_kernel(const float* __restrict__ X, const float* __restrict__ W,
                    unsigned short* __restrict__ Y) {
  __shared__ float red[8];
  __shared__ float rs;
  const int s = blockIdx.x, t = threadIdx.x;
  const float* xp = X + (size_t)s * DIM;
  const int base = t * 16;
  float ss = 0.f;
#pragma unroll
  for (int i = 0; i < 4; ++i) {
    v4f v = *(const v4f*)(xp + base + i * 4);
    ss += v[0]*v[0] + v[1]*v[1] + v[2]*v[2] + v[3]*v[3];
  }
  for (int m = 1; m < 32; m <<= 1) ss += __shfl_xor(ss, m);
  if ((t & 31) == 0) red[t >> 5] = ss;
  __syncthreads();
  if (t == 0) {
    float tot = 0.f;
#pragma unroll
    for (int i = 0; i < 8; ++i) tot += red[i];
    rs = rsqrtf(tot / (float)DIM + EPSF);
  }
  __syncthreads();
  const float r = rs;
#pragma unroll
  for (int i = 0; i < 16; ++i) {
    int c = base + i;
    Y[(size_t)s * DIM + c] = f32_to_bf16(xp[c] * r * W[c]);
  }
}

// =================================================================
// Tiled WMMA GEMM:  C[M,N] f32 = A[M,K]bf16 @ Bt[N,K]bf16^T (+Res)
// Block: 256 thr = 8 waves -> 128x128 C tile, K-step 64.
// Wave -> 32x64 (2x4 WMMA tiles), 16 WMMAs per K-step per wave.
// Both operands staged as pure b128 copies (no conversion in loop).
// =================================================================
__global__ __launch_bounds__(256)
void gemm_bf16_wmma(const unsigned short* __restrict__ A,   // [M,K] bf16
                    const unsigned short* __restrict__ Bt,  // [N,K] bf16
                    float* __restrict__ C,
                    const float* __restrict__ Res,
                    int M, int N, int K) {
  __shared__ unsigned short Asm[128][64];
  __shared__ unsigned short Bsm[128][64];
  const int tid  = threadIdx.x;
  const int m0   = blockIdx.y * 128;
  const int n0   = blockIdx.x * 128;
  const int wave = tid >> 5;
  const int lane = tid & 31;
  const int wm   = wave >> 1;          // 0..3  (M sub-tile)
  const int wn   = wave & 1;           // 0..1  (N sub-tile)
  const int sRow = tid >> 1;           // 0..127
  const int sCol = (tid & 1) * 32;     // 0 / 32
  const int lrow = lane & 15;
  const int lko  = (lane >> 4) * 8;    // 0 / 8

  v8f acc[2][4] = {};
  const unsigned short* ag = A  + (size_t)(m0 + sRow) * K + sCol;
  const unsigned short* bg = Bt + (size_t)(n0 + sRow) * K + sCol;

  for (int k0 = 0; k0 < K; k0 += 64) {
    // ---- stage A and Bt tiles: 4x b128 each, zero VALU ----
#pragma unroll
    for (int j = 0; j < 4; ++j) {
      *(v4u*)&Asm[sRow][sCol + 8 * j] = *(const v4u*)(ag + k0 + 8 * j);
      *(v4u*)&Bsm[sRow][sCol + 8 * j] = *(const v4u*)(bg + k0 + 8 * j);
    }
    if (k0 + 64 < K) {
      __builtin_prefetch((const void*)(ag + k0 + 64), 0, 0);
      __builtin_prefetch((const void*)(bg + k0 + 64), 0, 0);
    }
    __syncthreads();

#pragma unroll
    for (int kc = 0; kc < 2; ++kc) {
      FragB16 a[2];
#pragma unroll
      for (int ms = 0; ms < 2; ++ms) {
        const int ar = wm * 32 + ms * 16 + lrow;
        a[ms].q[0] = *(const v4u*)&Asm[ar][kc * 32 + lko];
        a[ms].q[1] = *(const v4u*)&Asm[ar][kc * 32 + lko + 16];
      }
#pragma unroll
      for (int ns = 0; ns < 4; ++ns) {
        FragB16 b;
        const int br = wn * 64 + ns * 16 + lrow;
        b.q[0] = *(const v4u*)&Bsm[br][kc * 32 + lko];
        b.q[1] = *(const v4u*)&Bsm[br][kc * 32 + lko + 16];
#pragma unroll
        for (int ms = 0; ms < 2; ++ms)
          acc[ms][ns] = __builtin_amdgcn_wmma_f32_16x16x32_bf16(
              false, a[ms].v, false, b.v, (short)0, acc[ms][ns], false, false);
      }
    }
    __syncthreads();
  }

  // ---- epilogue: C layout VGPR r -> row r (lanes 0-15) / r+8 (16-31)
  const int lcol  = lane & 15;
  const int rbase = (lane >> 4) * 8;
#pragma unroll
  for (int ms = 0; ms < 2; ++ms)
#pragma unroll
    for (int ns = 0; ns < 4; ++ns)
#pragma unroll
      for (int r = 0; r < 8; ++r) {
        int row = m0 + wm * 32 + ms * 16 + rbase + r;
        int col = n0 + wn * 64 + ns * 16 + lcol;
        float v = acc[ms][ns][r];
        if (Res != nullptr) v += Res[(size_t)row * N + col];
        C[(size_t)row * N + col] = v;
      }
}

// =================================================================
// RoPE + convert to bf16.  mode 0: Q (rope, *SCALE)  mode 1: K (rope)
// mode 2: V -> transposed [H][HD][S] for PV B-fragments
// =================================================================
__global__ __launch_bounds__(64)
void rope_cvt(const float* __restrict__ X, unsigned short* __restrict__ Y, int mode) {
  const int s = blockIdx.x, h = blockIdx.y, t = threadIdx.x;
  const float* xp = X + ((size_t)s * NH + h) * HD;
  if (mode == 2) {
#pragma unroll
    for (int d = t; d < HD; d += 64)
      Y[((size_t)h * HD + d) * SEQ + s] = f32_to_bf16(xp[d]);
    return;
  }
  float inv = __powf(10000.f, -2.f * (float)t / (float)HD);
  float ang = (float)s * inv;
  float sn, cs;
  __sincosf(ang, &sn, &cs);
  float x0 = xp[2 * t], x1 = xp[2 * t + 1];
  float r0 = x0 * cs - x1 * sn;
  float r1 = x0 * sn + x1 * cs;
  if (mode == 0) { r0 *= SCALEF; r1 *= SCALEF; }
  unsigned short* yp = Y + ((size_t)s * NH + h) * HD;
  yp[2 * t]     = f32_to_bf16(r0);
  yp[2 * t + 1] = f32_to_bf16(r1);
}

// =================================================================
// Flash attention, one wave per (head, 16-query tile).
// QK^T: 4 chained WMMA (K=32 over d) per 16-key tile; online softmax
// with half-wave shuffles; P re-laid out via 1KB LDS; PV: 8 WMMAs.
// =================================================================
__global__ __launch_bounds__(32)
void flash_attn_wmma(const unsigned short* __restrict__ Qb,  // [S,H,HD] bf16 (pre-scaled)
                     const unsigned short* __restrict__ Kb,  // [S,H,HD] bf16
                     const unsigned short* __restrict__ Vt,  // [H,HD,S] bf16
                     unsigned short* __restrict__ Ob) {      // [S, H*HD] bf16
  __shared__ unsigned short pbuf[16][32];
  const int lane  = threadIdx.x;
  const int q0    = blockIdx.x * 16;
  const int h     = blockIdx.y;
  const int lrow  = lane & 15;
  const int lhalf = lane >> 4;
  const int koff  = lhalf * 8;

  FragB16 qf[4];
  {
    const unsigned short* qp = Qb + ((size_t)(q0 + lrow) * NH + h) * HD;
#pragma unroll
    for (int c = 0; c < 4; ++c) {
      qf[c].q[0] = *(const v4u*)(qp + c * 32 + koff);
      qf[c].q[1] = *(const v4u*)(qp + c * 32 + koff + 16);
    }
  }

  v8f o[8] = {};
  float mr[8], lr[8];
#pragma unroll
  for (int r = 0; r < 8; ++r) { mr[r] = -1e30f; lr[r] = 0.f; }

  const int kend = q0 + 16;
  for (int kb = 0; kb < kend; kb += 32) {
    // ---- scores: two 16x16 tiles over a 32-key block ----
    v8f s[2] = {};
#pragma unroll
    for (int t = 0; t < 2; ++t) {
      const unsigned short* kp = Kb + ((size_t)(kb + t * 16 + lrow) * NH + h) * HD;
#pragma unroll
      for (int c = 0; c < 4; ++c) {
        FragB16 bfg;
        bfg.q[0] = *(const v4u*)(kp + c * 32 + koff);
        bfg.q[1] = *(const v4u*)(kp + c * 32 + koff + 16);
        s[t] = __builtin_amdgcn_wmma_f32_16x16x32_bf16(
            false, qf[c].v, false, bfg.v, (short)0, s[t], false, false);
      }
    }
    // ---- causal mask (only blocks straddling the diagonal) ----
    if (kb + 31 > q0) {
#pragma unroll
      for (int t = 0; t < 2; ++t) {
        int kcol = kb + t * 16 + lrow;
#pragma unroll
        for (int r = 0; r < 8; ++r) {
          int qrow = q0 + lhalf * 8 + r;
          if (kcol > qrow) s[t][r] = -1e30f;
        }
      }
    }
    // ---- online softmax: rows live in 16-lane halves ----
    float alpha[8];
#pragma unroll
    for (int r = 0; r < 8; ++r) {
      float v = fmaxf(s[0][r], s[1][r]);
      v = fmaxf(v, __shfl_xor(v, 1));
      v = fmaxf(v, __shfl_xor(v, 2));
      v = fmaxf(v, __shfl_xor(v, 4));
      v = fmaxf(v, __shfl_xor(v, 8));
      float mnew = fmaxf(mr[r], v);
      alpha[r] = __expf(mr[r] - mnew);
      mr[r] = mnew;
      float p0 = __expf(s[0][r] - mnew);
      float p1 = __expf(s[1][r] - mnew);
      s[0][r] = p0; s[1][r] = p1;
      float rs = p0 + p1;
      rs += __shfl_xor(rs, 1);
      rs += __shfl_xor(rs, 2);
      rs += __shfl_xor(rs, 4);
      rs += __shfl_xor(rs, 8);
      lr[r] = lr[r] * alpha[r] + rs;
    }
    // ---- re-layout P (C-layout -> A-fragment) via LDS ----
    __syncthreads();
#pragma unroll
    for (int t = 0; t < 2; ++t)
#pragma unroll
      for (int r = 0; r < 8; ++r)
        pbuf[lhalf * 8 + r][t * 16 + lrow] = f32_to_bf16(s[t][r]);
    __syncthreads();
    FragB16 pf;
    pf.q[0] = *(const v4u*)&pbuf[lrow][koff];
    pf.q[1] = *(const v4u*)&pbuf[lrow][koff + 16];
    // ---- O = O*alpha + P @ V  (V transposed: contiguous key runs) ----
#pragma unroll
    for (int j = 0; j < 8; ++j) {
#pragma unroll
      for (int r = 0; r < 8; ++r) o[j][r] *= alpha[r];
      const unsigned short* vp =
          Vt + ((size_t)h * HD + j * 16 + lrow) * SEQ + kb + koff;
      FragB16 vf;
      vf.q[0] = *(const v4u*)(vp);
      vf.q[1] = *(const v4u*)(vp + 16);
      o[j] = __builtin_amdgcn_wmma_f32_16x16x32_bf16(
          false, pf.v, false, vf.v, (short)0, o[j], false, false);
    }
  }
  // ---- normalize, store bf16 row-major [S, D] ----
#pragma unroll
  for (int j = 0; j < 8; ++j)
#pragma unroll
    for (int r = 0; r < 8; ++r) {
      int row = q0 + lhalf * 8 + r;
      int col = h * HD + j * 16 + lrow;
      Ob[(size_t)row * DIM + col] = f32_to_bf16(o[j][r] / lr[r]);
    }
}

// =================================================================
// SwiGLU elementwise: silu(g) * u  -> bf16
// =================================================================
__global__ __launch_bounds__(256)
void silu_mul_kernel(const float* __restrict__ G, const float* __restrict__ U,
                     unsigned short* __restrict__ Y, int n) {
  int i = blockIdx.x * blockDim.x + threadIdx.x;
  if (i < n) {
    float g = G[i];
    float sg = g / (1.f + __expf(-g));
    Y[i] = f32_to_bf16(sg * U[i]);
  }
}

// =================================================================
// Host orchestration
// =================================================================
extern "C" void kernel_launch(void* const* d_in, const int* in_sizes, int n_in,
                              void* d_out, int out_size, void* d_ws, size_t ws_size,
                              hipStream_t stream) {
  const float* x      = (const float*)d_in[0];
  const float* ln_w   = (const float*)d_in[1];
  const float* ffln_w = (const float*)d_in[2];
  const float* wq     = (const float*)d_in[3];
  const float* wk     = (const float*)d_in[4];
  const float* wv     = (const float*)d_in[5];
  const float* wo     = (const float*)d_in[6];
  const float* wg     = (const float*)d_in[7];
  const float* w1     = (const float*)d_in[8];
  const float* w2     = (const float*)d_in[9];
  float* out = (float*)d_out;

  char* ws = (char*)d_ws;
  size_t off = 0;
  const size_t DD_BF = (size_t)DIM * DIM * 2;      // 33.55 MB
  const size_t DH_BF = (size_t)DIM * HID * 2;      // 90.18 MB
  const size_t SD_BF = (size_t)SEQ * DIM * 2;      // 16.78 MB
  const size_t SD_F  = (size_t)SEQ * DIM * 4;      // 33.55 MB
  const size_t SH_F  = (size_t)SEQ * HID * 4;      // 90.18 MB
  const size_t SH_BF = (size_t)SEQ * HID * 2;      // 45.09 MB

  // weights, pre-transposed bf16: Bt[N][K]
  unsigned short* wqT = (unsigned short*)(ws + off); off += DD_BF;
  unsigned short* wkT = (unsigned short*)(ws + off); off += DD_BF;
  unsigned short* wvT = (unsigned short*)(ws + off); off += DD_BF;
  unsigned short* woT = (unsigned short*)(ws + off); off += DD_BF;
  unsigned short* wgT = (unsigned short*)(ws + off); off += DH_BF;
  unsigned short* w1T = (unsigned short*)(ws + off); off += DH_BF;
  unsigned short* w2T = (unsigned short*)(ws + off); off += DH_BF;
  // activations
  unsigned short* hbf   = (unsigned short*)(ws + off); off += SD_BF;
  float*          qf    = (float*)(ws + off);          off += SD_F;
  float*          kf    = (float*)(ws + off);          off += SD_F;
  float*          vf    = (float*)(ws + off);          off += SD_F;
  unsigned short* qbf   = (unsigned short*)(ws + off); off += SD_BF;
  unsigned short* kbf   = (unsigned short*)(ws + off); off += SD_BF;
  unsigned short* vtbf  = (unsigned short*)(ws + off); off += SD_BF;
  unsigned short* obf   = (unsigned short*)(ws + off); off += SD_BF;
  float*          x1    = (float*)(ws + off);          off += SD_F;
  float*          gf    = qf;  // reuse q/k/v f32 region (100.7MB >= 90.2MB)
  float*          uf    = (float*)(ws + off);          off += SH_F;
  unsigned short* gated = (unsigned short*)(ws + off); off += SH_BF;

  // ---- one-time weight convert+transpose (read f32 once) ----
  transpose_cvt_kernel<<<dim3(DIM/64, DIM/64), 256, 0, stream>>>(wq, wqT, DIM, DIM);
  transpose_cvt_kernel<<<dim3(DIM/64, DIM/64), 256, 0, stream>>>(wk, wkT, DIM, DIM);
  transpose_cvt_kernel<<<dim3(DIM/64, DIM/64), 256, 0, stream>>>(wv, wvT, DIM, DIM);
  transpose_cvt_kernel<<<dim3(DIM/64, DIM/64), 256, 0, stream>>>(wo, woT, DIM, DIM);
  transpose_cvt_kernel<<<dim3(HID/64, DIM/64), 256, 0, stream>>>(wg, wgT, DIM, HID);
  transpose_cvt_kernel<<<dim3(HID/64, DIM/64), 256, 0, stream>>>(w1, w1T, DIM, HID);
  transpose_cvt_kernel<<<dim3(DIM/64, HID/64), 256, 0, stream>>>(w2, w2T, HID, DIM);

  const dim3 gemmBlk(256);
  const dim3 gD(DIM / 128, SEQ / 128);   // N=4096 GEMMs
  const dim3 gH(HID / 128, SEQ / 128);   // N=11008 GEMMs (86 x 16)

  // ---- attention sublayer ----
  rmsnorm_kernel<<<SEQ, 256, 0, stream>>>(x, ln_w, hbf);
  gemm_bf16_wmma<<<gD, gemmBlk, 0, stream>>>(hbf, wqT, qf, nullptr, SEQ, DIM, DIM);
  gemm_bf16_wmma<<<gD, gemmBlk, 0, stream>>>(hbf, wkT, kf, nullptr, SEQ, DIM, DIM);
  gemm_bf16_wmma<<<gD, gemmBlk, 0, stream>>>(hbf, wvT, vf, nullptr, SEQ, DIM, DIM);
  rope_cvt<<<dim3(SEQ, NH), 64, 0, stream>>>(qf, qbf, 0);
  rope_cvt<<<dim3(SEQ, NH), 64, 0, stream>>>(kf, kbf, 1);
  rope_cvt<<<dim3(SEQ, NH), 64, 0, stream>>>(vf, vtbf, 2);
  flash_attn_wmma<<<dim3(SEQ / 16, NH), 32, 0, stream>>>(qbf, kbf, vtbf, obf);
  gemm_bf16_wmma<<<gD, gemmBlk, 0, stream>>>(obf, woT, x1, x, SEQ, DIM, DIM);

  // ---- SwiGLU MLP sublayer ----
  rmsnorm_kernel<<<SEQ, 256, 0, stream>>>(x1, ffln_w, hbf);
  gemm_bf16_wmma<<<gH, gemmBlk, 0, stream>>>(hbf, wgT, gf, nullptr, SEQ, HID, DIM);
  gemm_bf16_wmma<<<gH, gemmBlk, 0, stream>>>(hbf, w1T, uf, nullptr, SEQ, HID, DIM);
  {
    int n = SEQ * HID;
    silu_mul_kernel<<<(n + 255) / 256, 256, 0, stream>>>(gf, uf, gated, n);
  }
  gemm_bf16_wmma<<<gD, gemmBlk, 0, stream>>>(gated, w2T, out, x1, SEQ, DIM, HID);
}
